// ConditionalTripletLoss_70145405878277
// MI455X (gfx1250) — matrix operand order repaired and checked by reading, not compiled, not run on hardware
//
#include <hip/hip_runtime.h>
#include <hip/hip_bf16.h>
#include <math.h>

typedef __attribute__((ext_vector_type(16))) __bf16 v16bf;
typedef __attribute__((ext_vector_type(8)))  float  v8f;

#define B_ROWS 131072
#define D_DIM  256
#define C_CLS  128
#define LSTR   130            // LDS logits row stride (floats); 130%64=2 -> conflict-free column scan
#define NPART  (B_ROWS / 8)   // loss partials (one per 256-thread WG, 8 rows each)

// ---------------------------------------------------------------------------
// K1: logits = E @ W^T + b via v_wmma_f32_16x16x32_bf16, then top-2 -> target_neg
// 128 threads (4 waves). Each wave: 16 rows x 128 cols, K=256 in 8 steps of 32.
// ---------------------------------------------------------------------------
__global__ __launch_bounds__(128)
void k_logits_top2(const float* __restrict__ E, const float* __restrict__ W,
                   const float* __restrict__ bvec, const int* __restrict__ labels,
                   int* __restrict__ target_neg)
{
    __shared__ float lg[64 * LSTR];

    const int tid  = threadIdx.x;
    const int wave = tid >> 5;
    const int lane = tid & 31;
    const int half = lane >> 4;     // 0: lanes 0-15, 1: lanes 16-31
    const int l16  = lane & 15;
    const long rowbase = (long)blockIdx.x * 64;

    // ---- Preload A: row (rowbase + wave*16 + l16), all K, bf16, WMMA A layout.
    // lanes 0-15: v0..3 = K 0..7, v4..7 = K 16..23 ; lanes 16-31: K 8..15, 24..31.
    const int arow = (int)rowbase + wave * 16 + l16;
    const float* __restrict__ erow = E + (long)arow * D_DIM;
    v16bf A[8];
#pragma unroll
    for (int kk = 0; kk < 8; ++kk) {
        const int kb = kk * 32 + half * 8;
#pragma unroll
        for (int e = 0; e < 8; ++e) {
            A[kk][e]     = (__bf16)erow[kb + e];
            A[kk][e + 8] = (__bf16)erow[kb + 16 + e];
        }
    }

    // ---- 8 column tiles of 16 classes each
#pragma unroll
    for (int nt = 0; nt < 8; ++nt) {
        const int cls = nt * 16 + l16;                 // this lane's column (class)
        const float* __restrict__ wrow = W + (long)cls * D_DIM;
        v8f acc = {};
#pragma unroll
        for (int kk = 0; kk < 8; ++kk) {
            // B layout (32x16 bf16): lanes 0-15 hold K 0..7, 16..23 of their column,
            // lanes 16-31 hold K 8..15, 24..31; column n of B = row n of W.
            const int kb = kk * 32 + half * 8;
            v16bf Bm;
#pragma unroll
            for (int e = 0; e < 8; ++e) { Bm[e] = (__bf16)wrow[kb + e]; Bm[e + 8] = (__bf16)wrow[kb + 16 + e]; }
            acc = __builtin_amdgcn_wmma_f32_16x16x32_bf16(
                      false, A[kk], false, Bm, (short)0, acc, false, false);
        }
        const float bias = bvec[cls];
        // C/D layout: lane holds column cls, VGPR j -> row j + half*8
#pragma unroll
        for (int j = 0; j < 8; ++j) {
            const int mloc = wave * 16 + half * 8 + j;
            lg[mloc * LSTR + cls] = acc[j] + bias;
        }
    }
    __syncthreads();

    // ---- top-2 per row (JAX top_k tie-break: lowest index wins on ties)
    if (tid < 64) {
        const int row = (int)rowbase + tid;
        float v1 = -INFINITY, v2 = -INFINITY;
        int   i1 = 0, i2 = 0;
        for (int c = 0; c < C_CLS; ++c) {
            const float v = lg[tid * LSTR + c];
            if (v > v1)      { v2 = v1; i2 = i1; v1 = v; i1 = c; }
            else if (v > v2) { v2 = v;  i2 = c; }
        }
        const int lab = labels[row];
        target_neg[row] = (i1 == lab) ? i2 : i1;
    }
}

// ---------------------------------------------------------------------------
__global__ __launch_bounds__(256)
void k_zero(int* __restrict__ p, int n)
{
    const int i = blockIdx.x * 256 + threadIdx.x;
    if (i < n) p[i] = 0;
}

// K2: class histogram (LDS-staged integer atomics -> deterministic)
__global__ __launch_bounds__(256)
void k_count(const int* __restrict__ labels, int* __restrict__ counts)
{
    __shared__ int h[C_CLS];
    for (int c = threadIdx.x; c < C_CLS; c += 256) h[c] = 0;
    __syncthreads();
    const int stride = gridDim.x * 256;
    for (int i = blockIdx.x * 256 + threadIdx.x; i < B_ROWS; i += stride)
        atomicAdd(&h[labels[i]], 1);
    __syncthreads();
    for (int c = threadIdx.x; c < C_CLS; c += 256) atomicAdd(&counts[c], h[c]);
}

// K3: exclusive prefix sum over 128 counts (1 block, trivial)
__global__ __launch_bounds__(128)
void k_scan(const int* __restrict__ counts, int* __restrict__ starts)
{
    const int t = threadIdx.x;
    int s = 0;
    for (int c = 0; c < t; ++c) s += counts[c];
    starts[t] = s;
}

// K4: stable counting sort: one WG per class -> rank[i], order[starts[c]+rank]=i
__global__ __launch_bounds__(256)
void k_rank(const int* __restrict__ labels, const int* __restrict__ starts,
            int* __restrict__ rank, int* __restrict__ order)
{
    const int c = blockIdx.x;
    __shared__ int wtot[8];
    const int tid = threadIdx.x, wave = tid >> 5, lane = tid & 31;
    const int s0 = starts[c];
    int base = 0;
    for (int i0 = 0; i0 < B_ROWS; i0 += 256) {
        const int i = i0 + tid;
        const int flag = (labels[i] == c) ? 1 : 0;
        const unsigned long long m = __ballot(flag);
        const int wpre = __popcll(m & ((1ull << lane) - 1ull));
        if (lane == 0) wtot[wave] = __popcll(m);
        __syncthreads();
        int wbase = 0, tot = 0;
        for (int w = 0; w < 8; ++w) { const int v = wtot[w]; if (w < wave) wbase += v; tot += v; }
        if (flag) {
            const int r = base + wbase + wpre;
            rank[i] = r;
            order[s0 + r] = i;
        }
        base += tot;
        __syncthreads();
    }
}

// K5: one wave per row: sample pos/neg, distances, hinge; per-WG partials
__global__ __launch_bounds__(256)
void k_loss(const float* __restrict__ E, const int* __restrict__ labels,
            const int* __restrict__ counts, const int* __restrict__ starts,
            const int* __restrict__ rank, const int* __restrict__ order,
            const int* __restrict__ target_neg,
            const int* __restrict__ r_pos, const int* __restrict__ r_neg,
            float* __restrict__ ploss, int* __restrict__ pvalid)
{
    __shared__ float sl[8];
    __shared__ int   sv[8];
    const int tid = threadIdx.x, wave = tid >> 5, lane = tid & 31;
    const int i = blockIdx.x * 8 + wave;

    const int lab = labels[i];
    const int np  = counts[lab] - 1;
    const int rk  = rank[i];
    int pr = r_pos[i] % max(np, 1);
    pr += (pr >= rk) ? 1 : 0;
    const int pidx = order[min(starts[lab] + pr, B_ROWS - 1)];

    const int tn = target_neg[i];
    const int nn = counts[tn];
    const int nr = r_neg[i] % max(nn, 1);
    const int nidx = order[min(starts[tn] + nr, B_ROWS - 1)];

    const float* __restrict__ ei = E + (long)i    * D_DIM;
    const float* __restrict__ ep = E + (long)pidx * D_DIM;
    const float* __restrict__ en = E + (long)nidx * D_DIM;

    float dp = 0.f, dn = 0.f;
#pragma unroll
    for (int e = 0; e < 8; ++e) {
        const int k = lane * 8 + e;
        const float x = ei[k];
        const float a = x - ep[k]; dp += a * a;
        const float b = x - en[k]; dn += b * b;
    }
#pragma unroll
    for (int m = 16; m >= 1; m >>= 1) {
        dp += __shfl_xor(dp, m, 32);
        dn += __shfl_xor(dn, m, 32);
    }
    if (lane == 0) {
        const int valid = (np > 0) && (nn > 0);
        float L = sqrtf(dp) - sqrtf(dn) + 1.0f;   // MARGIN = 1.0
        L = valid ? fmaxf(L, 0.f) : 0.f;
        sl[wave] = L;
        sv[wave] = valid;
    }
    __syncthreads();
    if (tid == 0) {
        float s = 0.f; int v = 0;
        for (int w = 0; w < 8; ++w) { s += sl[w]; v += sv[w]; }
        ploss[blockIdx.x]  = s;
        pvalid[blockIdx.x] = v;
    }
}

// K6: deterministic final reduction (fixed per-thread order + fixed tree)
__global__ __launch_bounds__(256)
void k_final(const float* __restrict__ ploss, const int* __restrict__ pvalid,
             int nparts, float* __restrict__ out)
{
    __shared__ float sf[256];
    __shared__ int   si[256];
    const int t = threadIdx.x;
    float s = 0.f; int v = 0;
    for (int i = t; i < nparts; i += 256) { s += ploss[i]; v += pvalid[i]; }
    sf[t] = s; si[t] = v;
    __syncthreads();
    for (int off = 128; off >= 1; off >>= 1) {
        if (t < off) { sf[t] += sf[t + off]; si[t] += si[t + off]; }
        __syncthreads();
    }
    if (t == 0) out[0] = sf[0] / (float)max(si[0], 1);
}

// ---------------------------------------------------------------------------
extern "C" void kernel_launch(void* const* d_in, const int* in_sizes, int n_in,
                              void* d_out, int out_size, void* d_ws, size_t ws_size,
                              hipStream_t stream)
{
    const float* E      = (const float*)d_in[0];   // embeddings (B,D)
    const int*   labels = (const int*)  d_in[1];   // (B,)
    const float* W      = (const float*)d_in[2];   // (C,D)
    const float* bvec   = (const float*)d_in[3];   // (C,)
    const int*   r_pos  = (const int*)  d_in[4];   // (B,)
    const int*   r_neg  = (const int*)  d_in[5];   // (B,)
    float*       out    = (float*)d_out;

    char* ws = (char*)d_ws;
    int*   target_neg = (int*)ws;   ws += (size_t)B_ROWS * sizeof(int);
    int*   counts     = (int*)ws;   ws += (size_t)C_CLS  * sizeof(int);
    int*   starts     = (int*)ws;   ws += (size_t)C_CLS  * sizeof(int);
    int*   rank       = (int*)ws;   ws += (size_t)B_ROWS * sizeof(int);
    int*   order      = (int*)ws;   ws += (size_t)B_ROWS * sizeof(int);
    float* ploss      = (float*)ws; ws += (size_t)NPART  * sizeof(float);
    int*   pvalid     = (int*)ws;

    k_zero      <<<1,            256, 0, stream>>>(counts, C_CLS);
    k_logits_top2<<<B_ROWS / 64, 128, 0, stream>>>(E, W, bvec, labels, target_neg);
    k_count     <<<512,          256, 0, stream>>>(labels, counts);
    k_scan      <<<1,            128, 0, stream>>>(counts, starts);
    k_rank      <<<C_CLS,        256, 0, stream>>>(labels, starts, rank, order);
    k_loss      <<<B_ROWS / 8,   256, 0, stream>>>(E, labels, counts, starts, rank, order,
                                                   target_neg, r_pos, r_neg, ploss, pvalid);
    k_final     <<<1,            256, 0, stream>>>(ploss, pvalid, NPART, out);
}
